// LCSA_84224308674682
// MI455X (gfx1250) — compile-verified, matrix operand order checked by reading, not run on hardware
//
#include <hip/hip_runtime.h>
#include <hip/hip_bf16.h>

// ---------------- problem constants ----------------
#define BB      2
#define LL      2048
#define DM      512
#define NHEADS  8
#define HEADW   64
#define KTAPS   5
#define NPOS    (BB * LL)            // 4096
#define NQ      (NHEADS * HEADW)     // 512
#define NKV     (NHEADS * KTAPS * HEADW) // 2560
#define NALL    (NQ + NKV + NKV)     // 5632  (q | k | v)

// ---------------- vector types ----------------
typedef __attribute__((ext_vector_type(16))) __bf16          v16bf;
typedef __attribute__((ext_vector_type(8)))  float           v8f;
typedef __attribute__((ext_vector_type(8)))  unsigned short  v8us;

union FragAB { v16bf bf; v8us us[2]; };

static __device__ __forceinline__ unsigned short f2bf(float f) {
    union { float f; unsigned u; } a; a.f = f;
    unsigned u = a.u;
    u += 0x7fffu + ((u >> 16) & 1u);   // round-to-nearest-even
    return (unsigned short)(u >> 16);
}

// 16-byte global -> LDS async copy via GLOBAL_LOAD_ASYNC_TO_LDS_B128 (ASYNCcnt,
// no VGPR round trip). Inline asm bypasses the builtin's addrspace typing.
static __device__ __forceinline__ void copy16_g2lds(unsigned short* lds,
                                                    const unsigned short* g) {
    unsigned       ldsOff = (unsigned)(uintptr_t)lds;
    unsigned long long ga = (unsigned long long)(uintptr_t)g;
    asm volatile("global_load_async_to_lds_b128 %0, %1, off"
                 :: "v"(ldsOff), "v"(ga)
                 : "memory");
}

static __device__ __forceinline__ void wait_async_le4() {
    asm volatile("s_wait_asynccnt 0x4" ::: "memory");
}
static __device__ __forceinline__ void wait_async_0() {
    asm volatile("s_wait_asynccnt 0x0" ::: "memory");
}

// ---------------- pack kernel: f32 -> bf16 weights/activations ----------------
__global__ __launch_bounds__(256) void pack_kernel(
    const float* __restrict__ x,
    const float* __restrict__ qw, const float* __restrict__ kw, const float* __restrict__ vw,
    const float* __restrict__ qb, const float* __restrict__ kb, const float* __restrict__ vb,
    const float* __restrict__ ow,
    unsigned short* __restrict__ Xbf,    // NPOS x DM
    unsigned short* __restrict__ Wbf,    // NALL x DM   (rows: q 0..511, k 512..3071, v 3072..5631)
    unsigned short* __restrict__ OWbf,   // DM x NQ
    float* __restrict__ biasAll)         // NALL
{
    const long long xN  = (long long)NPOS * DM;
    const long long wN  = (long long)NALL * DM;
    const long long owN = (long long)DM * NQ;
    const long long total = xN + wN + owN + NALL;
    for (long long i = (long long)blockIdx.x * blockDim.x + threadIdx.x;
         i < total; i += (long long)gridDim.x * blockDim.x) {
        if (i < xN) {
            Xbf[i] = f2bf(x[i]);
        } else if (i < xN + wN) {
            long long j = i - xN;
            int row = (int)(j >> 9);
            int col = (int)(j & 511);
            float s;
            if (row < NQ)            s = qw[(long long)row * DM + col];
            else if (row < NQ + NKV) s = kw[(long long)(row - NQ) * DM + col];
            else                     s = vw[(long long)(row - NQ - NKV) * DM + col];
            Wbf[j] = f2bf(s);
        } else if (i < xN + wN + owN) {
            long long j = i - xN - wN;
            OWbf[j] = f2bf(ow[j]);
        } else {
            int j = (int)(i - xN - wN - owN);
            float s;
            if (j < NQ)            s = qb[j];
            else if (j < NQ + NKV) s = kb[j - NQ];
            else                   s = vb[j - NQ - NKV];
            biasAll[j] = s;
        }
    }
}

// ---------------- bf16 WMMA GEMM:  Out[M x N] = A[M x K] * B[N x K]^T + bias[N] ----------------
// Block 128x128, BK=32, 8 waves; wave -> 32(M) x 64(N) = 8 x v_wmma_f32_16x16x32_bf16 / K-step.
// Double-buffered LDS staged with GLOBAL_LOAD_ASYNC_TO_LDS_B128 (ASYNCcnt pipelining).
// LDS rows padded to 56 halves (112 B): 16B-aligned b128 ops, bank stride 28 words
// -> conflict-free across the 16 rows of a fragment load.
template<int BM, int BN, int BK>
__global__ __launch_bounds__(256) void gemm_bf16(
    const unsigned short* __restrict__ A,   // M x K, row-major, ld = K
    const unsigned short* __restrict__ Bm,  // N x K, row-major, ld = K
    const float* __restrict__ bias,         // N
    float* __restrict__ Out,                // M x ldOut
    int K, int ldOut)
{
    constexpr int LDSK = BK + 24;           // 56 halves = 112 bytes per row
    __shared__ __align__(16) unsigned short As[2][BM * LDSK];
    __shared__ __align__(16) unsigned short Bs[2][BN * LDSK];

    const int tid   = threadIdx.x;
    const int lane  = tid & 31;
    const int wave  = tid >> 5;        // 0..7
    const int waveM = wave >> 1;       // 0..3  -> 32-row slab
    const int waveN = wave & 1;        // 0..1  -> 64-col slab
    const int r15   = lane & 15;
    const int hi    = lane >> 4;       // selects K-half per ISA 7.12.2 layout
    const int blockM = blockIdx.y * BM;
    const int blockN = blockIdx.x * BN;

    // per-thread staging coordinates: 2 chunks of 8 halves for A, same for B
    int rowA[2], colA[2];
#pragma unroll
    for (int i = 0; i < 2; ++i) {
        int chunk = tid * 2 + i;             // 0..511
        rowA[i] = chunk >> 2;                // BK/8 = 4 chunks per row
        colA[i] = (chunk & 3) * 8;
    }

    v8f acc[2][4];
#pragma unroll
    for (int mt = 0; mt < 2; ++mt)
#pragma unroll
        for (int nt = 0; nt < 4; ++nt)
            acc[mt][nt] = v8f{};

    const int nstep = K / BK;

    auto issue = [&](int buf, int k0) {
#pragma unroll
        for (int i = 0; i < 2; ++i) {
            copy16_g2lds(&As[buf][rowA[i] * LDSK + colA[i]],
                         A  + (size_t)(blockM + rowA[i]) * K + k0 + colA[i]);
            copy16_g2lds(&Bs[buf][rowA[i] * LDSK + colA[i]],
                         Bm + (size_t)(blockN + rowA[i]) * K + k0 + colA[i]);
        }
    };

    auto compute = [&](int buf) {
        // fragment loads per ISA 7.12.2 (lanes 0-15: K 0..7 & 16..23; 16-31: 8..15 & 24..31)
        FragAB aF[2], bF[4];
#pragma unroll
        for (int mt = 0; mt < 2; ++mt) {
            int row = waveM * 32 + mt * 16 + r15;
            aF[mt].us[0] = *(const v8us*)(&As[buf][row * LDSK + hi * 8]);
            aF[mt].us[1] = *(const v8us*)(&As[buf][row * LDSK + hi * 8 + 16]);
        }
#pragma unroll
        for (int nt = 0; nt < 4; ++nt) {
            int row = waveN * 64 + nt * 16 + r15;
            bF[nt].us[0] = *(const v8us*)(&Bs[buf][row * LDSK + hi * 8]);
            bF[nt].us[1] = *(const v8us*)(&Bs[buf][row * LDSK + hi * 8 + 16]);
        }
#pragma unroll
        for (int mt = 0; mt < 2; ++mt)
#pragma unroll
            for (int nt = 0; nt < 4; ++nt)
                acc[mt][nt] = __builtin_amdgcn_wmma_f32_16x16x32_bf16(
                    false, aF[mt].bf, false, bF[nt].bf,
                    (short)0, acc[mt][nt], false, false);
    };

    // prologue: stage step 0 into buffer 0 (4 async b128 per wave)
    issue(0, 0);

    // steady state: branch-free pipeline
    for (int it = 0; it < nstep - 1; ++it) {
        const int cur = it & 1;
        issue(cur ^ 1, (it + 1) * BK);   // next tile streams into other buffer
        wait_async_le4();                // current buffer's 4 ops complete
        __syncthreads();                 // all waves staged current buffer
        compute(cur);
        __syncthreads();                 // all reads done before buffer rewrite
    }

    // epilogue step
    wait_async_0();
    __syncthreads();
    compute((nstep - 1) & 1);

    // ---- epilogue: C/D layout -> VGPR r = (M=r + hi*8, N=r15) ----
#pragma unroll
    for (int nt = 0; nt < 4; ++nt) {
        int n = blockN + waveN * 64 + nt * 16 + r15;
        float bv = bias[n];
#pragma unroll
        for (int mt = 0; mt < 2; ++mt) {
            int mBase = blockM + waveM * 32 + mt * 16 + hi * 8;
#pragma unroll
            for (int vr = 0; vr < 8; ++vr)
                Out[(size_t)(mBase + vr) * ldOut + n] = acc[mt][nt][vr] + bv;
        }
    }
}

// ---------------- attention: one wave per (position, head) ----------------
// Y layout per row (5632): [ q: h*64+o | 512 + (h*5+kk)*64+o | 3072 + (h*5+kk)*64+o ]
__global__ __launch_bounds__(256) void attn_kernel(
    const float* __restrict__ Y,
    const float* __restrict__ kbias,   // 8*5*64
    const float* __restrict__ vbias,   // 8*5*64
    unsigned short* __restrict__ attnOut)  // NPOS x 512 bf16
{
    const int lane = threadIdx.x & 31;
    const int w    = (int)((blockIdx.x * blockDim.x + threadIdx.x) >> 5);
    const int pos  = w >> 3;
    const int h    = w & 7;
    if (pos >= NPOS) return;
    const int b = pos >> 11;
    const int l = pos & (LL - 1);
    const int dil = 1 << (h & 3);      // dilations {1,2,4,8,1,2,4,8}

    const size_t rowQ = (size_t)pos * NALL + h * HEADW;
    const float q0 = Y[rowQ + lane];
    const float q1 = Y[rowQ + lane + 32];

    float logit[KTAPS];
#pragma unroll
    for (int kk = 0; kk < KTAPS; ++kk) {
        int lp = l + (kk - 2) * dil;
        float k0v, k1v;
        if (lp >= 0 && lp < LL) {
            size_t rk = (size_t)(b * LL + lp) * NALL + NQ + (h * KTAPS + kk) * HEADW;
            k0v = Y[rk + lane];
            k1v = Y[rk + lane + 32];
        } else {
            int o = (h * KTAPS + kk) * HEADW;
            k0v = kbias[o + lane];
            k1v = kbias[o + lane + 32];
        }
        float p = q0 * k0v + q1 * k1v;
#pragma unroll
        for (int s = 16; s >= 1; s >>= 1)
            p += __shfl_xor(p, s, 32);
        logit[kk] = p;
    }

    float m = logit[0];
#pragma unroll
    for (int kk = 1; kk < KTAPS; ++kk) m = fmaxf(m, logit[kk]);
    float e[KTAPS], s = 0.f;
#pragma unroll
    for (int kk = 0; kk < KTAPS; ++kk) { e[kk] = __expf(logit[kk] - m); s += e[kk]; }
    const float inv = 1.f / s;

    float a0 = 0.f, a1 = 0.f;
#pragma unroll
    for (int kk = 0; kk < KTAPS; ++kk) {
        float wgt = e[kk] * inv;
        int lp = l + (kk - 2) * dil;
        float v0, v1;
        if (lp >= 0 && lp < LL) {
            size_t rv = (size_t)(b * LL + lp) * NALL + NQ + NKV + (h * KTAPS + kk) * HEADW;
            v0 = Y[rv + lane];
            v1 = Y[rv + lane + 32];
        } else {
            int o = (h * KTAPS + kk) * HEADW;
            v0 = vbias[o + lane];
            v1 = vbias[o + lane + 32];
        }
        a0 += wgt * v0;
        a1 += wgt * v1;
    }
    a0 *= 0.125f;  // 1/sqrt(64)
    a1 *= 0.125f;

    size_t ro = (size_t)pos * NQ + h * HEADW;
    attnOut[ro + lane]      = f2bf(a0);
    attnOut[ro + lane + 32] = f2bf(a1);
}

// ---------------- host launcher ----------------
extern "C" void kernel_launch(void* const* d_in, const int* in_sizes, int n_in,
                              void* d_out, int out_size, void* d_ws, size_t ws_size,
                              hipStream_t stream) {
    const float* x   = (const float*)d_in[0];
    const float* q_w = (const float*)d_in[1];
    const float* q_b = (const float*)d_in[2];
    const float* k_w = (const float*)d_in[3];
    const float* k_b = (const float*)d_in[4];
    const float* v_w = (const float*)d_in[5];
    const float* v_b = (const float*)d_in[6];
    const float* o_w = (const float*)d_in[7];
    const float* o_b = (const float*)d_in[8];
    float* out = (float*)d_out;

    // workspace layout (bytes)
    char* ws = (char*)d_ws;
    const size_t OFF_XBF  = 0;
    const size_t OFF_WBF  = OFF_XBF + (size_t)NPOS * DM * 2;
    const size_t OFF_OWBF = OFF_WBF + (size_t)NALL * DM * 2;
    const size_t OFF_BIAS = OFF_OWBF + (size_t)DM * NQ * 2;
    const size_t OFF_Y    = (OFF_BIAS + (size_t)NALL * 4 + 255) & ~(size_t)255;
    const size_t OFF_ATTN = OFF_Y + (size_t)NPOS * NALL * 4;

    unsigned short* Xbf  = (unsigned short*)(ws + OFF_XBF);
    unsigned short* Wbf  = (unsigned short*)(ws + OFF_WBF);
    unsigned short* OWbf = (unsigned short*)(ws + OFF_OWBF);
    float*          bAll = (float*)(ws + OFF_BIAS);
    float*          Y    = (float*)(ws + OFF_Y);
    unsigned short* attn = (unsigned short*)(ws + OFF_ATTN);

    // 1) pack to bf16
    pack_kernel<<<2048, 256, 0, stream>>>(x, q_w, k_w, v_w, q_b, k_b, v_b, o_w,
                                          Xbf, Wbf, OWbf, bAll);

    // 2) fused QKV projection GEMM: Y(4096 x 5632) = Xbf @ Wbf^T + biasAll
    gemm_bf16<128, 128, 32><<<dim3(NALL / 128, NPOS / 128), 256, 0, stream>>>(
        Xbf, Wbf, bAll, Y, DM, NALL);

    // 3) dilated local attention over precomputed q/k/v rows
    attn_kernel<<<(NPOS * NHEADS * 32) / 256, 256, 0, stream>>>(Y, k_b, v_b, attn);

    // 4) output projection GEMM: out(4096 x 512) = attn @ OWbf^T + o_b
    gemm_bf16<128, 128, 32><<<dim3(NQ / 128, NPOS / 128), 256, 0, stream>>>(
        attn, OWbf, o_b, out, NQ, NQ);
}
